// Generator_9174050144373
// MI455X (gfx1250) — compile-verified, matrix operand order checked by reading, not run on hardware
//
#include <hip/hip_runtime.h>
#include <cstddef>
#include <cstdint>

// ---------------- problem constants ----------------
#define H     100
#define H2X   200
#define H3    300
#define T     512
#define OUTV  128000
#define SOS   0
#define EOS   1

// WMMA GEMV tiling: 128000 rows -> 8000 tiles of 16 rows; K 100 -> padded 128 -> 4 chunks of 32
#define NTILES 8000
#define NCH    4
#define NB4    200          // blocks for logits kernel
#define TPB4   256          // 8 waves
#define TILES_PER_WAVE  5
#define TILES_PER_BLOCK 40  // 8 waves * 5

typedef __attribute__((ext_vector_type(16))) _Float16 v16h;
typedef __attribute__((ext_vector_type(8)))  float    v8f;

// ---------------- workspace layout (bytes) ----------------
#define OFF_W16    0u                         // f16 swizzled out_W: 8000*4*32*16 halves = 32,768,000 B
#define OFF_ENC    32768000u                  // enc_outs: T*H floats = 204,800 B
#define OFF_HN     (OFF_ENC + 204800u)        // hN: 100 floats (pad 512)
#define OFF_H      (OFF_HN + 512u)            // h state: 100 floats (pad 512)
#define OFF_H2     (OFF_H + 512u)             // h2: 100 floats (pad 512)
#define OFF_BFRAG  (OFF_H2 + 512u)            // B fragments: 4 chunks * 2 lane-halves * 16 halves = 256 B
#define OFF_EMIT   (OFF_BFRAG + 256u)         // emitted tokens: T ints = 2048 B
#define OFF_STATE  (OFF_EMIT + 2048u)         // [0]=tok [1]=done (pad 256)
#define OFF_LOGZ   (OFF_STATE + 256u)         // log-sum-exp per step: T floats = 2048 B
#define OFF_PV     (OFF_LOGZ + 2048u)         // per-block max val: NB4 floats (pad 1024)
#define OFF_PI     (OFF_PV + 1024u)           // per-block argmax: NB4 ints (pad 1024)
#define OFF_PS     (OFF_PI + 1024u)           // per-block sum-exp: NB4 floats (pad 1024)
#define WS_NEED    (OFF_PS + 1024u)

// A-matrix 16x32 f16 per-lane K map from the CDNA5 ISA table (lane-half a, packed idx j)
__device__ __forceinline__ int kmap(int a, int j) {
    int v = j >> 1, hb = j & 1;
    return (v < 4) ? (a * 8 + 2 * v + hb) : (16 + a * 8 + 2 * (v - 4) + hb);
}

// ---------------- 1) swizzle out_W fp32 -> f16 in WMMA A layout ----------------
__global__ void swz_kernel(const float* __restrict__ outW, char* __restrict__ ws) {
    _Float16* W16 = (_Float16*)(ws + OFF_W16);
    const size_t total = (size_t)NTILES * NCH * 32 * 16;
    for (size_t e = (size_t)blockIdx.x * blockDim.x + threadIdx.x; e < total;
         e += (size_t)gridDim.x * blockDim.x) {
        int j     = (int)(e & 15);
        int lane  = (int)((e >> 4) & 31);
        int chunk = (int)((e >> 9) & 3);
        int tile  = (int)(e >> 11);
        int m = lane & 15, a = lane >> 4;
        int k = chunk * 32 + kmap(a, j);
        int row = tile * 16 + m;
        float val = (k < H) ? outW[(size_t)row * H + k] : 0.0f;
        W16[e] = (_Float16)val;
    }
}

// ---------------- 2) encoder: 512 sequential GRU steps, one block ----------------
__global__ void enc_kernel(const int* __restrict__ a, const float* __restrict__ emb,
                           const float* __restrict__ Wih, const float* __restrict__ Whh,
                           const float* __restrict__ bih, const float* __restrict__ bhh,
                           char* __restrict__ ws) {
    __shared__ float sx[H], sh[H], sgi[H3], sgh[H3];
    float* enc_outs = (float*)(ws + OFF_ENC);
    float* hN = (float*)(ws + OFF_HN);
    const int tid = threadIdx.x; // 512 threads
    if (tid < H) sh[tid] = 0.0f;
    __syncthreads();
    for (int t = 0; t < T; ++t) {
        int tok = a[t];
        if (tid < H) sx[tid] = emb[(size_t)tok * H + tid];
        __syncthreads();
        for (int r = tid; r < 2 * H3; r += 512) {
            if (r < H3) {
                const float* w = Wih + (size_t)r * H;
                float acc = bih[r];
                for (int c = 0; c < H; ++c) acc += w[c] * sx[c];
                sgi[r] = acc;
            } else {
                int rr = r - H3;
                const float* w = Whh + (size_t)rr * H;
                float acc = bhh[rr];
                for (int c = 0; c < H; ++c) acc += w[c] * sh[c];
                sgh[rr] = acc;
            }
        }
        __syncthreads();
        if (tid < H) {
            float r = 1.0f / (1.0f + expf(-(sgi[tid] + sgh[tid])));
            float z = 1.0f / (1.0f + expf(-(sgi[tid + H] + sgh[tid + H])));
            float n = tanhf(sgi[tid + 2 * H] + r * sgh[tid + 2 * H]);
            float hv = (1.0f - z) * n + z * sh[tid];
            enc_outs[t * H + tid] = hv;
            sh[tid] = hv;
        }
        __syncthreads();
    }
    if (tid < H) hN[tid] = sh[tid];
}

// ---------------- 3) per-step glue: reduce prev argmax/sumexp, advance state,
//                    attention + context + combine + GRU -> h2 + B fragment ----------------
__global__ void step_kernel(int t,
                            const float* __restrict__ dec_emb, const float* __restrict__ attn_W,
                            const float* __restrict__ attn_b, const float* __restrict__ comb_W,
                            const float* __restrict__ comb_b, const float* __restrict__ Wih,
                            const float* __restrict__ Whh, const float* __restrict__ bih,
                            const float* __restrict__ bhh, char* __restrict__ ws) {
    __shared__ float sx[H], sh[H], sctx[H], scomb[H], sh2[H], sgi[H3], sgh[H3];
    __shared__ float sscore[T];
    __shared__ float red[512], redsum[512];
    __shared__ int   redi[512];
    __shared__ int   sDone, sTok;
    __shared__ float smax, ssum;

    float* h_ws  = (float*)(ws + OFF_H);
    float* h2_ws = (float*)(ws + OFF_H2);
    float* hN    = (float*)(ws + OFF_HN);
    _Float16* bf = (_Float16*)(ws + OFF_BFRAG);
    int*   emit  = (int*)(ws + OFF_EMIT);
    int*   state = (int*)(ws + OFF_STATE);
    float* logz  = (float*)(ws + OFF_LOGZ);
    float* pv    = (float*)(ws + OFF_PV);
    int*   pi    = (int*)(ws + OFF_PI);
    float* ps    = (float*)(ws + OFF_PS);
    const float* enc_outs = (const float*)(ws + OFF_ENC);
    const int tid = threadIdx.x; // 512 threads

    if (t == 0) {
        if (tid < H) h_ws[tid] = hN[tid];
        if (tid == 0) { state[0] = SOS; state[1] = 0; sDone = 0; sTok = SOS; }
    } else {
        // deterministic fixed-order reduction of previous step's partials
        float v, s; int ix;
        if (tid < NB4) { v = pv[tid]; ix = pi[tid]; s = ps[tid]; }
        else           { v = -3.0e38f; ix = 0x7fffffff; s = 0.0f; }
        red[tid] = v; redi[tid] = ix; redsum[tid] = s;
        __syncthreads();
        for (int sd = 256; sd > 0; sd >>= 1) {
            if (tid < sd) {
                float v2 = red[tid + sd]; int i2 = redi[tid + sd];
                if (v2 > red[tid] || (v2 == red[tid] && i2 < redi[tid])) {
                    red[tid] = v2; redi[tid] = i2;
                }
                redsum[tid] += redsum[tid + sd];
            }
            __syncthreads();
        }
        if (tid == 0) {
            int nxt = redi[0];
            logz[t - 1] = logf(redsum[0]);                 // log-sum-exp of step t-1 logits
            int olddone = state[1];
            emit[t - 1] = olddone ? EOS : nxt;             // emit uses OLD done
            int ndone = olddone | (nxt == EOS);
            int ntok  = olddone ? EOS : nxt;
            state[0] = ntok; state[1] = ndone;
            sDone = olddone; sTok = ntok;
        }
        __syncthreads();
        if (!sDone && tid < H) h_ws[tid] = h2_ws[tid];     // h <- h2 unless frozen
    }
    __syncthreads();
    if (t >= T) return;                                    // tail launch: state-only

    // ---- forward for step t ----
    const int tok = sTok;
    if (tid < H) { sx[tid] = dec_emb[(size_t)tok * H + tid]; sh[tid] = h_ws[tid]; }
    __syncthreads();

    { // attention scores: row per thread (T == 512 == blockDim)
        const float* w = attn_W + (size_t)tid * H2X;
        float acc = attn_b[tid];
        for (int c = 0; c < H; ++c) acc += w[c] * sx[c];
        for (int c = 0; c < H; ++c) acc += w[H + c] * sh[c];
        sscore[tid] = acc;
    }
    __syncthreads();
    // softmax (stable)
    red[tid] = sscore[tid];
    __syncthreads();
    for (int sd = 256; sd > 0; sd >>= 1) {
        if (tid < sd) red[tid] = fmaxf(red[tid], red[tid + sd]);
        __syncthreads();
    }
    if (tid == 0) smax = red[0];
    __syncthreads();
    float e = expf(sscore[tid] - smax);
    sscore[tid] = e;                  // own slot only
    red[tid] = e;
    __syncthreads();
    for (int sd = 256; sd > 0; sd >>= 1) {
        if (tid < sd) red[tid] += red[tid + sd];
        __syncthreads();
    }
    if (tid == 0) ssum = red[0];
    __syncthreads();
    // context
    if (tid < H) {
        float acc = 0.0f;
        for (int tt = 0; tt < T; ++tt) acc += sscore[tt] * enc_outs[tt * H + tid];
        sctx[tid] = acc / ssum;
    }
    __syncthreads();
    // combine + relu
    if (tid < H) {
        const float* w = comb_W + (size_t)tid * H2X;
        float acc = comb_b[tid];
        for (int c = 0; c < H; ++c) acc += w[c] * sx[c];
        for (int c = 0; c < H; ++c) acc += w[H + c] * sctx[c];
        scomb[tid] = fmaxf(acc, 0.0f);
    }
    __syncthreads();
    // GRU gates
    for (int r = tid; r < 2 * H3; r += 512) {
        if (r < H3) {
            const float* w = Wih + (size_t)r * H;
            float acc = bih[r];
            for (int c = 0; c < H; ++c) acc += w[c] * scomb[c];
            sgi[r] = acc;
        } else {
            int rr = r - H3;
            const float* w = Whh + (size_t)rr * H;
            float acc = bhh[rr];
            for (int c = 0; c < H; ++c) acc += w[c] * sh[c];
            sgh[rr] = acc;
        }
    }
    __syncthreads();
    if (tid < H) {
        float r = 1.0f / (1.0f + expf(-(sgi[tid] + sgh[tid])));
        float z = 1.0f / (1.0f + expf(-(sgi[tid + H] + sgh[tid + H])));
        float n = tanhf(sgi[tid + 2 * H] + r * sgh[tid + 2 * H]);
        float hv = (1.0f - z) * n + z * sh[tid];
        sh2[tid] = hv;
        h2_ws[tid] = hv;
    }
    __syncthreads();
    // B fragment: h2 replicated across all 16 N columns, packed per the A-mirrored K map
    if (tid < 128) {
        int c = tid >> 5, rem = tid & 31, a = rem >> 4, j = rem & 15;
        int k = c * 32 + kmap(a, j);
        float val = (k < H) ? sh2[k] : 0.0f;
        bf[(c * 2 + a) * 16 + j] = (_Float16)val;
    }
}

// ---------------- 4) logits GEMV via WMMA + partial argmax/sum-exp ----------------
__global__ __launch_bounds__(TPB4) void logits_kernel(int t, const float* __restrict__ out_b,
                                                      char* __restrict__ ws,
                                                      float* __restrict__ d_out) {
    const _Float16* W16 = (const _Float16*)(ws + OFF_W16);
    const _Float16* bf  = (const _Float16*)(ws + OFF_BFRAG);
    float* pv = (float*)(ws + OFF_PV);
    int*   pi = (int*)(ws + OFF_PI);
    float* ps = (float*)(ws + OFF_PS);
    float* logits = d_out + T + (size_t)t * OUTV;

    const int tid = threadIdx.x, lane = tid & 31, wave = tid >> 5, a = lane >> 4;

    v16h B[NCH];
#pragma unroll
    for (int c = 0; c < NCH; ++c) B[c] = *(const v16h*)(bf + (c * 2 + a) * 16);

    float lsum = 0.0f, bmax = -3.0e38f; int bidx = 0x7fffffff;

#pragma unroll 1
    for (int q = 0; q < TILES_PER_WAVE; ++q) {
        const int tile = blockIdx.x * TILES_PER_BLOCK + wave * TILES_PER_WAVE + q;
        const _Float16* abase = W16 + (((size_t)tile * NCH) * 32 + lane) * 16;
        v16h A[NCH];
#pragma unroll
        for (int c = 0; c < NCH; ++c) A[c] = *(const v16h*)(abase + (size_t)c * 32 * 16);
        if (q + 1 < TILES_PER_WAVE)
            __builtin_prefetch(abase + (size_t)NCH * 32 * 16, 0, 1);  // global_prefetch_b8

        v8f acc = {};
#pragma unroll
        for (int c = 0; c < NCH; ++c)
            acc = __builtin_amdgcn_wmma_f32_16x16x32_f16(false, A[c], false, B[c],
                                                         (short)0, acc, false, false);

        // column 0 of D = tile @ h2; lane 0 holds rows 0-7, lane 16 rows 8-15
        if ((lane & 15) == 0) {
            const int rowbase = tile * 16 + a * 8;
            float vals[8];
#pragma unroll
            for (int i = 0; i < 8; ++i) vals[i] = acc[i] + out_b[rowbase + i];
            *(float4*)(logits + rowbase)     = make_float4(vals[0], vals[1], vals[2], vals[3]);
            *(float4*)(logits + rowbase + 4) = make_float4(vals[4], vals[5], vals[6], vals[7]);
#pragma unroll
            for (int i = 0; i < 8; ++i) {
                float vv = vals[i];
                lsum += expf(vv);
                if (vv > bmax) { bmax = vv; bidx = rowbase + i; } // first-max (rows increasing)
            }
        }
    }

    __shared__ float rv[TPB4], rs[TPB4];
    __shared__ int   ri[TPB4];
    rv[tid] = bmax; ri[tid] = bidx; rs[tid] = lsum;
    __syncthreads();
    for (int sd = TPB4 / 2; sd > 0; sd >>= 1) {
        if (tid < sd) {
            float v2 = rv[tid + sd]; int i2 = ri[tid + sd];
            if (v2 > rv[tid] || (v2 == rv[tid] && i2 < ri[tid])) { rv[tid] = v2; ri[tid] = i2; }
            rs[tid] += rs[tid + sd];
        }
        __syncthreads();
    }
    if (tid == 0) { pv[blockIdx.x] = rv[0]; pi[blockIdx.x] = ri[0]; ps[blockIdx.x] = rs[0]; }
}

// ---------------- 5) finalize: log_softmax normalization + token output ----------------
__global__ void norm_kernel(char* __restrict__ ws, float* __restrict__ d_out) {
    const float* logz = (const float*)(ws + OFF_LOGZ);
    const int*   emit = (const int*)(ws + OFF_EMIT);
    const uint32_t total = (uint32_t)T * (uint32_t)OUTV;
    const uint32_t stride = gridDim.x * blockDim.x;
    for (uint32_t e = blockIdx.x * blockDim.x + threadIdx.x; e < total; e += stride) {
        uint32_t tt = e / OUTV;
        d_out[T + (size_t)e] -= logz[tt];
    }
    if (blockIdx.x == 0)
        for (int i = threadIdx.x; i < T; i += blockDim.x) d_out[i] = (float)emit[i];
}

// ---------------- host ----------------
extern "C" void kernel_launch(void* const* d_in, const int* in_sizes, int n_in,
                              void* d_out, int out_size, void* d_ws, size_t ws_size,
                              hipStream_t stream) {
    (void)in_sizes; (void)n_in; (void)out_size;
    if (ws_size < (size_t)WS_NEED) return;

    const int*   a_tensor = (const int*)d_in[0];
    const float* enc_emb  = (const float*)d_in[1];
    const float* enc_Wih  = (const float*)d_in[2];
    const float* enc_Whh  = (const float*)d_in[3];
    const float* enc_bih  = (const float*)d_in[4];
    const float* enc_bhh  = (const float*)d_in[5];
    const float* dec_emb  = (const float*)d_in[6];
    const float* attn_W   = (const float*)d_in[7];
    const float* attn_b   = (const float*)d_in[8];
    const float* comb_W   = (const float*)d_in[9];
    const float* comb_b   = (const float*)d_in[10];
    const float* dec_Wih  = (const float*)d_in[11];
    const float* dec_Whh  = (const float*)d_in[12];
    const float* dec_bih  = (const float*)d_in[13];
    const float* dec_bhh  = (const float*)d_in[14];
    const float* out_W    = (const float*)d_in[15];
    const float* out_b    = (const float*)d_in[16];

    char*  ws  = (char*)d_ws;
    float* out = (float*)d_out;

    // one-time per call: f16-swizzle out_W (L2-resident for all 512 steps) + encoder scan
    swz_kernel<<<2048, 256, 0, stream>>>(out_W, ws);
    enc_kernel<<<1, 512, 0, stream>>>(a_tensor, enc_emb, enc_Wih, enc_Whh, enc_bih, enc_bhh, ws);

    for (int t = 0; t < T; ++t) {
        step_kernel<<<1, 512, 0, stream>>>(t, dec_emb, attn_W, attn_b, comb_W, comb_b,
                                           dec_Wih, dec_Whh, dec_bih, dec_bhh, ws);
        logits_kernel<<<NB4, TPB4, 0, stream>>>(t, out_b, ws, out);
    }
    // tail: consume step 511 partials (emit[511], logz[511])
    step_kernel<<<1, 512, 0, stream>>>(T, dec_emb, attn_W, attn_b, comb_W, comb_b,
                                       dec_Wih, dec_Whh, dec_bih, dec_bhh, ws);
    norm_kernel<<<2048, 256, 0, stream>>>(ws, out);
}
